// WindowAttention_3_branches_68375879352952
// MI455X (gfx1250) — matrix operand order repaired and checked by reading, not compiled, not run on hardware
//
#include <hip/hip_runtime.h>

typedef __attribute__((ext_vector_type(16))) _Float16 v16h;
typedef __attribute__((ext_vector_type(8)))  _Float16 v8h;
typedef __attribute__((ext_vector_type(8)))  float    v8f;

#define NWIN  49
#define NPAD  64
#define DIM   128
#define NHEAD 4
#define HDIM  32

// ---------------------------------------------------------------------------
// WMMA helpers (v_wmma_f32_16x16x32_f16: D = A(16x32 f16) * B(32x16 f16) + C)
// ---------------------------------------------------------------------------
__device__ __forceinline__ v8f wmma_f16(v16h a, v16h b, v8f c) {
  return __builtin_amdgcn_wmma_f32_16x16x32_f16(
      /*neg_a=*/false, a, /*neg_b=*/false, b,
      /*c_mod=*/(short)0, c, /*reuse_a=*/false, /*reuse_b=*/false);
}

// A fragment, 16x32 f16, from row-major [rows][stride] storage.
// Lane L (0..15): M=L, holds K = col_off+0..15
// Lane L (16..31): M=L-16, holds K = col_off+16..31
__device__ __forceinline__ v16h load_a_frag(const _Float16* base, int stride,
                                            int row_off, int col_off) {
  int lane = threadIdx.x & 31;
  const _Float16* p = base + (row_off + (lane & 15)) * stride
                           + col_off + ((lane >> 4) << 4);
  v8h lo = *(const v8h*)p;
  v8h hi = *(const v8h*)(p + 8);
  v16h out;
#pragma unroll
  for (int i = 0; i < 8; ++i) { out[i] = lo[i]; out[i + 8] = hi[i]; }
  return out;
}

// B fragment, 32x16 f16 (K x N), from storage where B[k][n] = M[n][k]
// (M row-major [n][stride]).  Lanes 0..15: N=lane, K = k_off+0..15;
// lanes 16..31: N=lane-16, K = k_off+16..31 (contiguous per lane).
__device__ __forceinline__ v16h load_b_frag(const _Float16* base, int stride,
                                            int n_off, int k_off) {
  int lane = threadIdx.x & 31;
  const _Float16* p = base + (n_off + (lane & 15)) * stride
                           + k_off + ((lane >> 4) << 4);
  v8h lo = *(const v8h*)p;
  v8h hi = *(const v8h*)(p + 8);
  v16h out;
#pragma unroll
  for (int i = 0; i < 8; ++i) { out[i] = lo[i]; out[i + 8] = hi[i]; }
  return out;
}

// Prepacked weight B-fragment from global workspace: [tile][lane][16] f16
__device__ __forceinline__ v16h load_w_frag(const _Float16* w, int tileIdx) {
  int lane = threadIdx.x & 31;
  const _Float16* p = w + ((size_t)tileIdx * 32 + lane) * 16;
  v8h lo = *(const v8h*)p;
  v8h hi = *(const v8h*)(p + 8);
  v16h out;
#pragma unroll
  for (int i = 0; i < 8; ++i) { out[i] = lo[i]; out[i + 8] = hi[i]; }
  return out;
}

// ---------------------------------------------------------------------------
// Prep kernel: pack weights into f16 B-fragment order; materialize padded
// relative-position bias in C-fragment order (f32), -1e30 on padded cols.
// ---------------------------------------------------------------------------
#define WQKV_ELEMS (24 * 4 * 32 * 16)   // 49152 f16
#define WPROJ_ELEMS (8 * 4 * 32 * 16)   // 16384 f16
#define WBIAS_ELEMS (4 * 4 * 4 * 8 * 32) // 16384 f32

__global__ void prep_kernel(const float* __restrict__ qkv_w,
                            const float* __restrict__ proj_w,
                            const float* __restrict__ bias_table,
                            _Float16* __restrict__ wQKV,
                            _Float16* __restrict__ wPROJ,
                            float* __restrict__ wBIAS) {
  int i = blockIdx.x * blockDim.x + threadIdx.x;
  if (i < WQKV_ELEMS) {
    int nt = i / 2048, rem = i % 2048;
    int ks = rem / 512, l = rem % 512;
    int lane = l / 16, hh = l % 16;
    int n = nt * 16 + (lane & 15);
    int k = ks * 32 + ((lane >> 4) << 4) + hh;
    wQKV[i] = (_Float16)qkv_w[n * DIM + k];      // B[k][n] = qkv_w[n][k]
  } else if (i < WQKV_ELEMS + WPROJ_ELEMS) {
    int j = i - WQKV_ELEMS;
    int nt = j / 2048, rem = j % 2048;
    int ks = rem / 512, l = rem % 512;
    int lane = l / 16, hh = l % 16;
    int n = nt * 16 + (lane & 15);
    int k = ks * 32 + ((lane >> 4) << 4) + hh;
    wPROJ[j] = (_Float16)proj_w[n * DIM + k];
  } else if (i < WQKV_ELEMS + WPROJ_ELEMS + WBIAS_ELEMS) {
    int t = i - WQKV_ELEMS - WPROJ_ELEMS;
    int lane = t & 31;
    int r = (t >> 5) & 7;
    int nt = (t >> 8) & 3;
    int mt = (t >> 10) & 3;
    int h = t >> 12;
    int row = mt * 16 + r + ((lane >> 4) << 3);  // query token
    int col = nt * 16 + (lane & 15);             // key token
    float v;
    if (row >= NWIN || col >= NWIN) {
      v = -1e30f;
    } else {
      int ih = row / 7, iw = row % 7, jh = col / 7, jw = col % 7;
      int ridx = (ih - jh + 6) * 13 + (iw - jw + 6);
      v = bias_table[ridx * NHEAD + h];
    }
    wBIAS[t] = v;
  }
}

// ---------------------------------------------------------------------------
// Per-head: scores = Q*K2^T (+bias as WMMA C operand) -> softmax -> probs f16
// ---------------------------------------------------------------------------
__device__ __forceinline__ void compute_probs(const _Float16* Qh,
                                              const _Float16* K2h,
                                              const float* __restrict__ wBIAS,
                                              _Float16* Ph, int h) {
  int lane = threadIdx.x & 31;
  v16h bk[4];
#pragma unroll
  for (int nt = 0; nt < 4; ++nt) bk[nt] = load_b_frag(K2h, HDIM, nt * 16, 0);
#pragma unroll
  for (int mt = 0; mt < 4; ++mt) {
    v16h af = load_a_frag(Qh, HDIM, mt * 16, 0);
    v8f acc[4];
#pragma unroll
    for (int nt = 0; nt < 4; ++nt) {
      v8f cb;
      const float* bp = wBIAS + (size_t)(((h * 4 + mt) * 4 + nt) * 8) * 32 + lane;
#pragma unroll
      for (int r = 0; r < 8; ++r) cb[r] = bp[r * 32];
      acc[nt] = wmma_f16(af, bk[nt], cb);
    }
    int row0 = mt * 16 + ((lane >> 4) << 3);
    int coll = lane & 15;
#pragma unroll
    for (int r = 0; r < 8; ++r) {
      // Row-wise softmax: reduce over 4 nt tiles and 16 lanes of this half.
      float mx = fmaxf(fmaxf(acc[0][r], acc[1][r]), fmaxf(acc[2][r], acc[3][r]));
      mx = fmaxf(mx, __shfl_xor(mx, 1, 32));
      mx = fmaxf(mx, __shfl_xor(mx, 2, 32));
      mx = fmaxf(mx, __shfl_xor(mx, 4, 32));
      mx = fmaxf(mx, __shfl_xor(mx, 8, 32));
      float e0 = __expf(acc[0][r] - mx);
      float e1 = __expf(acc[1][r] - mx);
      float e2 = __expf(acc[2][r] - mx);
      float e3 = __expf(acc[3][r] - mx);
      float s = e0 + e1 + e2 + e3;
      s += __shfl_xor(s, 1, 32);
      s += __shfl_xor(s, 2, 32);
      s += __shfl_xor(s, 4, 32);
      s += __shfl_xor(s, 8, 32);
      float inv = 1.0f / s;
      int row = row0 + r;
      Ph[row * 64 +  0 + coll] = (_Float16)(e0 * inv);
      Ph[row * 64 + 16 + coll] = (_Float16)(e1 * inv);
      Ph[row * 64 + 32 + coll] = (_Float16)(e2 * inv);
      Ph[row * 64 + 48 + coll] = (_Float16)(e3 * inv);
    }
  }
}

// Per-head: O[:, h*32 .. h*32+31] = probs(64x64) * V(64x32), V stored
// transposed [feat 32][tok 64] so B-fragments are contiguous loads.
__device__ __forceinline__ void attn_v(const _Float16* Ph, const _Float16* Vth,
                                       _Float16* sO, int h) {
  int lane = threadIdx.x & 31;
  v16h bf[2][2];
#pragma unroll
  for (int ntf = 0; ntf < 2; ++ntf)
#pragma unroll
    for (int kt = 0; kt < 2; ++kt)
      bf[ntf][kt] = load_b_frag(Vth, NPAD, ntf * 16, kt * 32);
#pragma unroll
  for (int mt = 0; mt < 4; ++mt) {
    v16h a0 = load_a_frag(Ph, 64, mt * 16, 0);
    v16h a1 = load_a_frag(Ph, 64, mt * 16, 32);
#pragma unroll
    for (int ntf = 0; ntf < 2; ++ntf) {
      v8f acc = {};
      acc = wmma_f16(a0, bf[ntf][0], acc);
      acc = wmma_f16(a1, bf[ntf][1], acc);
      int row0 = mt * 16 + ((lane >> 4) << 3);
      int col = h * HDIM + ntf * 16 + (lane & 15);
#pragma unroll
      for (int r = 0; r < 8; ++r)
        sO[(row0 + r) * DIM + col] = (_Float16)acc[r];
    }
  }
}

// Projection: out(49x128) = O(64x128) * proj_w^T + proj_b.  Each wave owns
// 2 of the 8 column tiles.
__device__ __forceinline__ void proj_store(const _Float16* sO,
                                           const _Float16* __restrict__ wPROJ,
                                           const float* __restrict__ proj_b,
                                           float* __restrict__ outb, int wave) {
  int lane = threadIdx.x & 31;
#pragma unroll
  for (int i = 0; i < 2; ++i) {
    int nt = wave * 2 + i;
#pragma unroll
    for (int mt = 0; mt < 4; ++mt) {
      v8f acc = {};
#pragma unroll
      for (int ks = 0; ks < 4; ++ks) {
        v16h af = load_a_frag(sO, DIM, mt * 16, ks * 32);
        v16h bf = load_w_frag(wPROJ, nt * 4 + ks);
        acc = wmma_f16(af, bf, acc);
      }
      int col = nt * 16 + (lane & 15);
      float pb = proj_b[col];
      int row0 = mt * 16 + ((lane >> 4) << 3);
#pragma unroll
      for (int r = 0; r < 8; ++r) {
        int tok = row0 + r;
        if (tok < NWIN) outb[(size_t)tok * DIM + col] = acc[r] + pb;
      }
    }
  }
}

// ---------------------------------------------------------------------------
// Main kernel: one workgroup (4 waves) per window.
// ---------------------------------------------------------------------------
__global__ __launch_bounds__(128)
void win_attn_kernel(const float* __restrict__ x, const float* __restrict__ x2,
                     const float* __restrict__ qkv_b,
                     const float* __restrict__ proj_b,
                     const _Float16* __restrict__ wQKV,
                     const _Float16* __restrict__ wPROJ,
                     const float* __restrict__ wBIAS,
                     float* __restrict__ out, int B_) {
  extern __shared__ _Float16 smem[];
  _Float16* sX   = smem;              // [64][128]  x (f16)
  _Float16* sX2  = sX   + 8192;       // [64][128]
  _Float16* sQ   = sX2  + 8192;       // [4][64][32] q of x (pre-scaled)
  _Float16* sQ2  = sQ   + 8192;       // [4][64][32] q of x2 (pre-scaled)
  _Float16* sK2  = sQ2  + 8192;       // [4][64][32] k of x2
  _Float16* sVt  = sK2  + 8192;       // [4][32][64] v of x, transposed
  _Float16* sV2t = sVt  + 8192;       // [4][32][64] v of x2, transposed
  _Float16* sP   = sV2t + 8192;       // [4][64][64] probs
  _Float16* sO   = sP   + 16384;      // [64][128]  attn output staging

  const int b = blockIdx.x;
  const int tid = threadIdx.x;
  const int lane = tid & 31;
  const int wave = tid >> 5;
  const float SCALE = 0.17677669529663687f;  // 32^-0.5

  // ---- Phase 1: stage x, x2 into LDS as f16 (zero-pad rows 49..63) ----
  const float* xb  = x  + (size_t)b * NWIN * DIM;
  const float* x2b = x2 + (size_t)b * NWIN * DIM;
  for (int i = tid; i < NPAD * DIM; i += 128) {
    int row = i >> 7, col = i & 127;
    float v1 = (row < NWIN) ? xb[row * DIM + col] : 0.0f;
    float v2 = (row < NWIN) ? x2b[row * DIM + col] : 0.0f;
    sX[i]  = (_Float16)v1;
    sX2[i] = (_Float16)v2;
  }
  __syncthreads();

  // ---- Phase 2: QKV GEMMs  [64x128] x [128x384] for x and x2 ----
  // 192 output tiles (2 inputs * 4 mt * 24 nt) distributed over 4 waves.
  for (int t = wave; t < 192; t += 4) {
    int inp = t / 96;
    int rem = t - inp * 96;
    int mt = rem / 24;
    int nt = rem % 24;
    int qi = nt >> 3;  // 0=q 1=k 2=v
    if (inp == 0 && qi == 1) continue;  // k of x is never used
    const _Float16* A = inp ? sX2 : sX;
    v8f acc = {};
#pragma unroll
    for (int ks = 0; ks < 4; ++ks) {
      v16h af = load_a_frag(A, DIM, mt * 16, ks * 32);
      v16h bf = load_w_frag(wQKV, nt * 4 + ks);
      acc = wmma_f16(af, bf, acc);
    }
    int head = (nt >> 1) & 3;
    int fh = nt & 1;
    int coll = lane & 15;
    int row0 = mt * 16 + ((lane >> 4) << 3);
    int feat = fh * 16 + coll;
    float bias = qkv_b[nt * 16 + coll];
    if (qi == 0) {
      _Float16* dst = (inp ? sQ2 : sQ) + head * (NPAD * HDIM);
#pragma unroll
      for (int r = 0; r < 8; ++r)
        dst[(row0 + r) * HDIM + feat] = (_Float16)((acc[r] + bias) * SCALE);
    } else if (qi == 1) {  // inp==1 only
      _Float16* dst = sK2 + head * (NPAD * HDIM);
#pragma unroll
      for (int r = 0; r < 8; ++r)
        dst[(row0 + r) * HDIM + feat] = (_Float16)(acc[r] + bias);
    } else {               // V: store transposed [feat][tok]
      _Float16* dst = (inp ? sV2t : sVt) + head * (HDIM * NPAD);
#pragma unroll
      for (int r = 0; r < 8; ++r)
        dst[feat * NPAD + (row0 + r)] = (_Float16)(acc[r] + bias);
    }
  }
  __syncthreads();

  // Per-wave (per-head) views
  const _Float16* Qh  = sQ   + wave * (NPAD * HDIM);
  const _Float16* Q2h = sQ2  + wave * (NPAD * HDIM);
  const _Float16* K2h = sK2  + wave * (NPAD * HDIM);
  const _Float16* Vh  = sVt  + wave * (HDIM * NPAD);
  const _Float16* V2h = sV2t + wave * (HDIM * NPAD);
  _Float16* Ph = sP + wave * 4096;

  const size_t obase   = (size_t)b * NWIN * DIM;
  const size_t ostride = (size_t)B_ * NWIN * DIM;

  // ---- Phase 3a: attn2 = softmax(q2*k2^T*scale + bias) ----
  compute_probs(Q2h, K2h, wBIAS, Ph, wave);
  __syncthreads();

  // ---- out0 = proj(attn2 @ v) ----
  attn_v(Ph, Vh, sO, wave);
  __syncthreads();
  proj_store(sO, wPROJ, proj_b, out + obase, wave);
  __syncthreads();

  // ---- out1 = proj(attn2 @ v2) ----
  attn_v(Ph, V2h, sO, wave);
  __syncthreads();
  proj_store(sO, wPROJ, proj_b, out + ostride + obase, wave);
  __syncthreads();

  // ---- Phase 3b: attn3 = softmax(q*k2^T*scale + bias); out2 ----
  compute_probs(Qh, K2h, wBIAS, Ph, wave);
  __syncthreads();
  attn_v(Ph, V2h, sO, wave);
  __syncthreads();
  proj_store(sO, wPROJ, proj_b, out + 2 * ostride + obase, wave);
}

// ---------------------------------------------------------------------------
extern "C" void kernel_launch(void* const* d_in, const int* in_sizes, int n_in,
                              void* d_out, int out_size, void* d_ws,
                              size_t ws_size, hipStream_t stream) {
  const float* x          = (const float*)d_in[0];
  const float* x2         = (const float*)d_in[1];
  const float* qkv_w      = (const float*)d_in[2];
  const float* qkv_b      = (const float*)d_in[3];
  const float* proj_w     = (const float*)d_in[4];
  const float* proj_b     = (const float*)d_in[5];
  const float* bias_table = (const float*)d_in[6];
  float* out = (float*)d_out;

  int B_ = in_sizes[0] / (NWIN * DIM);

  _Float16* wQKV  = (_Float16*)d_ws;              // 49152 f16 (96 KB)
  _Float16* wPROJ = wQKV + WQKV_ELEMS;            // 16384 f16 (32 KB)
  float*    wBIAS = (float*)(wPROJ + WPROJ_ELEMS);// 16384 f32 (64 KB)

  int prep_items = WQKV_ELEMS + WPROJ_ELEMS + WBIAS_ELEMS;
  prep_kernel<<<(prep_items + 255) / 256, 256, 0, stream>>>(
      qkv_w, proj_w, bias_table, wQKV, wPROJ, wBIAS);

  size_t shmem_bytes = (8192 * 7 + 16384 + 8192) * sizeof(_Float16);  // 160 KB
  win_attn_kernel<<<B_, 128, shmem_bytes, stream>>>(
      x, x2, qkv_b, proj_b, wQKV, wPROJ, wBIAS, out, B_);
}
